// HunyuanImage3DecoderLayer_78469052498386
// MI455X (gfx1250) — matrix operand order repaired and checked by reading, not compiled, not run on hardware
//
#include <hip/hip_runtime.h>
#include <hip/hip_bf16.h>

// ---------------------------------------------------------------------------
// HunyuanImage3 decoder layer for MI455X (gfx1250), wave32 + WMMA bf16
// + async-to-LDS staging + TDM tensor_load_to_lds weight staging.
// ---------------------------------------------------------------------------
#define S_    1024
#define H_    1024
#define NH_   16
#define KVH_  8
#define HD_   64
#define G_    2
#define E_    64
#define TOPK_ 8
#define I_    3072
#define CAP_  512
#define NASGN (S_ * TOPK_)       // 8192 assignments
#define DUMMY_BASE NASGN         // scatter target for invalid rows
#define MOE_ROWS (NASGN + 64)

typedef __attribute__((ext_vector_type(16))) __bf16 v16bf;
typedef __attribute__((ext_vector_type(8)))  float  v8f;
typedef __attribute__((ext_vector_type(4)))  unsigned int u32x4;
typedef __attribute__((ext_vector_type(4)))  int i32x4_;
typedef __attribute__((ext_vector_type(8)))  int i32x8_;

union Frag { v16bf v; u32x4 u[2]; };

// A-operand fragment (16x32 bf16). rowbase points at a 64-byte row (32 bf16,
// k contiguous). lo lanes take K{0..7,16..23}, hi lanes K{8..15,24..31}.
__device__ __forceinline__ v16bf load_fragA(const char* rowbase, bool hi) {
  Frag f;
  const int o = hi ? 16 : 0;
  f.u[0] = *(const u32x4*)(rowbase + o);
  f.u[1] = *(const u32x4*)(rowbase + 32 + o);
  return f.v;
}
// B-operand fragment (32x16 bf16). colbase points at 64 bytes holding the 32
// K-values of this lane's column (N = lane&15). lo lanes K{0..7,16..23}, hi K{8..15,24..31}.
__device__ __forceinline__ v16bf load_fragB(const char* colbase, bool hi) {
  Frag f;
  const int o = hi ? 16 : 0;
  f.u[0] = *(const u32x4*)(colbase + o);
  f.u[1] = *(const u32x4*)(colbase + 32 + o);
  return f.v;
}

// --- CDNA5 async global->LDS copy (16B per lane), tracked by ASYNCcnt ------
__device__ __forceinline__ void async_ld_b128(void* lds_dst, const void* g) {
  const unsigned lds_off = (unsigned)(unsigned long long)lds_dst;  // AS(3) low bits
  asm volatile("global_load_async_to_lds_b128 %0, %1, off"
               :: "v"(lds_off), "v"((unsigned long long)g)
               : "memory");
}
__device__ __forceinline__ void wait_async0() {
  asm volatile("s_wait_asynccnt 0" ::: "memory");
}

// --- CDNA5 Tensor Data Mover: 2D fp32 tile (tile_cols x tile_rows), row
// stride row_stride_elems, DMA'd contiguously into LDS. D# per ISA §8.3/8.4.
__device__ __forceinline__ void tdm_load_f32_tile(void* lds_dst, const void* gsrc,
                                                  int tile_cols, int tile_rows,
                                                  int row_stride_elems) {
  const unsigned long long ga = (unsigned long long)gsrc;
  i32x4_ g0;
  g0[0] = 1;                                               // count=1 (valid D#)
  g0[1] = (int)(unsigned)(unsigned long long)lds_dst;      // lds_addr
  g0[2] = (int)(ga & 0xffffffffull);                       // global_addr lo
  g0[3] = (int)(((ga >> 32) & 0x01ffffffull) | 0x80000000u); // addr hi | type=2
  i32x8_ g1;
  g1[0] = 2 << 16;                                         // data_size = 4B
  g1[1] = (tile_cols & 0xffff) << 16;                      // tensor_dim0 lo16
  g1[2] = ((tile_cols >> 16) & 0xffff) |
          ((tile_rows & 0xffff) << 16);                    // dim0 hi | dim1 lo
  g1[3] = ((tile_rows >> 16) & 0xffff) |
          ((tile_cols & 0xffff) << 16);                    // dim1 hi | tile_dim0
  g1[4] = tile_rows & 0xffff;                              // tile_dim1, tile_dim2=0
  g1[5] = row_stride_elems;                                // tensor_dim0_stride lo32
  g1[6] = 0;
  g1[7] = 0;
  asm volatile("tensor_load_to_lds %0, %1" :: "s"(g0), "s"(g1) : "memory");
}

// ---------------------------------------------------------------------------
// RMSNorm over H, fp32 in -> bf16 (and optional fp32) out.  grid=S, block=256.
// ---------------------------------------------------------------------------
__global__ __launch_bounds__(256)
void rmsnorm_kernel(const float* __restrict__ x, const float* __restrict__ w,
                    __bf16* __restrict__ obf, float* __restrict__ of32) {
  const int t = blockIdx.x, tid = threadIdx.x;
  const float* row = x + (long)t * H_;
  float4 v = *(const float4*)(row + tid * 4);
  float ss = v.x * v.x + v.y * v.y + v.z * v.z + v.w * v.w;
  for (int m = 16; m >= 1; m >>= 1) ss += __shfl_xor(ss, m, 32);
  __shared__ float red[8];
  if ((tid & 31) == 0) red[tid >> 5] = ss;
  __syncthreads();
  ss = 0.f;
  for (int i = 0; i < 8; ++i) ss += red[i];
  const float inv = rsqrtf(ss / (float)H_ + 1e-6f);
  const float* pv = &v.x;
  for (int c = 0; c < 4; ++c) {
    const int i = tid * 4 + c;
    const float y = pv[c] * inv * w[i];
    obf[(long)t * H_ + i] = (__bf16)y;
    if (of32) of32[(long)t * H_ + i] = y;
  }
}

// ---------------------------------------------------------------------------
// Generic WMMA GEMM: C[f32] (+opt residual) = A[bf16] @ B[f32->bf16].
// Block 256 thr = 8 waves (4x2), tile 64x32, K-step 32.
// A tile: global_load_async_to_lds_b128 (works with row gather).
// B tile: TDM tensor_load_to_lds (fp32), then LDS convert to transposed bf16.
// ---------------------------------------------------------------------------
__global__ __launch_bounds__(256)
void gemm_bf16(const __bf16* __restrict__ A, int lda,
               const float* __restrict__ Bw, int ldb, long bExpStride,
               float* __restrict__ C, int ldc, int K,
               const float* __restrict__ residual,
               const int* __restrict__ a_idx, const int* __restrict__ c_idx,
               const int* __restrict__ cnt, int idx_stride, int c_dummy_base) {
  const int z  = blockIdx.z;
  const int m0 = blockIdx.y * 64;
  const int n0 = blockIdx.x * 32;
  const float* B = Bw + (long)z * bExpStride;
  const int mcount = cnt ? cnt[z] : 0x40000000;

  __shared__ __align__(16) __bf16 As[64][32];
  __shared__ __align__(16) float  Bs[32][32];    // raw fp32 tile (TDM dest)
  __shared__ __align__(16) __bf16 Bts[32][32];   // transposed bf16 [n][k]

  const int tid = threadIdx.x;
  const int lane = tid & 31;
  const int wid = __builtin_amdgcn_readfirstlane(tid >> 5);
  const int wm = wid >> 1, wn = wid & 1;
  const bool hi = lane >= 16;
  const int l15 = lane & 15;

  const int ar = tid >> 2;          // staging row 0..63
  const int ak = (tid & 3) * 8;     // staging k offset
  int arow;
  {
    const int g = m0 + ar;
    arow = a_idx ? ((g < mcount) ? a_idx[z * idx_stride + g] : 0) : g;
  }
  const int bk = tid >> 3;          // 0..31
  const int bn = (tid & 7) * 4;     // 0..28

  v8f acc = {};
  const int kTiles = K / 32;
  for (int kt = 0; kt < kTiles; ++kt) {
    const int k0 = kt * 32;
    // A tile: per-lane async copy into LDS (ASYNCcnt)
    async_ld_b128(&As[ar][ak], A + (long)arow * lda + k0 + ak);
    // B tile: one TDM descriptor per workgroup (TENSORcnt)
    if (wid == 0) {
      tdm_load_f32_tile(&Bs[0][0], B + (long)k0 * ldb + n0, 32, 32, ldb);
      __builtin_amdgcn_s_wait_tensorcnt(0);
    }
    wait_async0();
    __syncthreads();
    // convert fp32 LDS tile -> transposed bf16 fragment layout
    {
      const float4 f = *(const float4*)(&Bs[bk][bn]);
      Bts[bn + 0][bk] = (__bf16)f.x;
      Bts[bn + 1][bk] = (__bf16)f.y;
      Bts[bn + 2][bk] = (__bf16)f.z;
      Bts[bn + 3][bk] = (__bf16)f.w;
    }
    if (kt + 1 < kTiles)   // global_prefetch_b8: pull next tile toward L2
      __builtin_prefetch(B + (long)(k0 + 32 + bk) * ldb + n0, 0, 1);
    __syncthreads();
    const v16bf af = load_fragA((const char*)&As[wm * 16 + l15][0], hi);
    const v16bf bf = load_fragB((const char*)&Bts[wn * 16 + l15][0], hi);
    acc = __builtin_amdgcn_wmma_f32_16x16x32_bf16(false, af, false, bf,
                                                  (short)0, acc, false, false);
    __syncthreads();
  }

  const int n = n0 + wn * 16 + l15;
  for (int r = 0; r < 8; ++r) {
    const int gm = m0 + wm * 16 + r + (hi ? 8 : 0);
    int crow;
    if (c_idx) crow = (gm < mcount) ? c_idx[z * idx_stride + gm]
                                    : (c_dummy_base + (gm & 63));
    else       crow = gm;
    float v = acc[r];
    if (residual) v += residual[(long)gm * ldc + n];
    C[(long)crow * ldc + n] = v;
  }
}

// ---------------------------------------------------------------------------
// Fused GEMM + SwiGLU: act[.,c] = a * silu(b) where a = (X@Wgu)[.,c],
// b = (X@Wgu)[., I+c]. Dual B tiles (both halves) share the A fragments.
// A via async-to-LDS; both B tiles via TDM.  Shared MLP + MoE FFN1.
// ---------------------------------------------------------------------------
__global__ __launch_bounds__(256)
void ffn1_swiglu(const __bf16* __restrict__ A, int lda,
                 const float* __restrict__ Wgu, long bExpStride, int ldb,
                 __bf16* __restrict__ Act, int ldact, int K,
                 const int* __restrict__ a_idx, const int* __restrict__ c_idx,
                 const int* __restrict__ cnt, int idx_stride, int c_dummy_base) {
  const int z  = blockIdx.z;
  const int m0 = blockIdx.y * 64;
  const int n0 = blockIdx.x * 32;
  const float* B = Wgu + (long)z * bExpStride;
  const int mcount = cnt ? cnt[z] : 0x40000000;

  __shared__ __align__(16) __bf16 As[64][32];
  __shared__ __align__(16) float  Bsa[32][32];
  __shared__ __align__(16) float  Bsb[32][32];
  __shared__ __align__(16) __bf16 Bta[32][32];
  __shared__ __align__(16) __bf16 Btb[32][32];

  const int tid = threadIdx.x;
  const int lane = tid & 31;
  const int wid = __builtin_amdgcn_readfirstlane(tid >> 5);
  const int wm = wid >> 1, wn = wid & 1;
  const bool hi = lane >= 16;
  const int l15 = lane & 15;

  const int ar = tid >> 2;
  const int ak = (tid & 3) * 8;
  int arow;
  {
    const int g = m0 + ar;
    arow = a_idx ? ((g < mcount) ? a_idx[z * idx_stride + g] : 0) : g;
  }
  const int bk = tid >> 3;
  const int bn = (tid & 7) * 4;

  v8f acc_a = {}, acc_b = {};
  const int kTiles = K / 32;
  for (int kt = 0; kt < kTiles; ++kt) {
    const int k0 = kt * 32;
    async_ld_b128(&As[ar][ak], A + (long)arow * lda + k0 + ak);
    if (wid == 0) {
      tdm_load_f32_tile(&Bsa[0][0], B + (long)k0 * ldb + n0,       32, 32, ldb);
      tdm_load_f32_tile(&Bsb[0][0], B + (long)k0 * ldb + I_ + n0,  32, 32, ldb);
      __builtin_amdgcn_s_wait_tensorcnt(0);
    }
    wait_async0();
    __syncthreads();
    {
      const float4 fa = *(const float4*)(&Bsa[bk][bn]);
      const float4 fb = *(const float4*)(&Bsb[bk][bn]);
      Bta[bn + 0][bk] = (__bf16)fa.x; Bta[bn + 1][bk] = (__bf16)fa.y;
      Bta[bn + 2][bk] = (__bf16)fa.z; Bta[bn + 3][bk] = (__bf16)fa.w;
      Btb[bn + 0][bk] = (__bf16)fb.x; Btb[bn + 1][bk] = (__bf16)fb.y;
      Btb[bn + 2][bk] = (__bf16)fb.z; Btb[bn + 3][bk] = (__bf16)fb.w;
    }
    if (kt + 1 < kTiles)
      __builtin_prefetch(B + (long)(k0 + 32 + bk) * ldb + n0, 0, 1);
    __syncthreads();
    const v16bf af  = load_fragA((const char*)&As[wm * 16 + l15][0], hi);
    const v16bf bfa = load_fragB((const char*)&Bta[wn * 16 + l15][0], hi);
    const v16bf bfb = load_fragB((const char*)&Btb[wn * 16 + l15][0], hi);
    acc_a = __builtin_amdgcn_wmma_f32_16x16x32_bf16(false, af, false, bfa,
                                                    (short)0, acc_a, false, false);
    acc_b = __builtin_amdgcn_wmma_f32_16x16x32_bf16(false, af, false, bfb,
                                                    (short)0, acc_b, false, false);
    __syncthreads();
  }

  const int n = n0 + wn * 16 + l15;
  for (int r = 0; r < 8; ++r) {
    const int gm = m0 + wm * 16 + r + (hi ? 8 : 0);
    int crow;
    if (c_idx) crow = (gm < mcount) ? c_idx[z * idx_stride + gm]
                                    : (c_dummy_base + (gm & 63));
    else       crow = gm;
    const float a = acc_a[r], b = acc_b[r];
    const float act = a * (b / (1.f + __expf(-b)));
    Act[(long)crow * ldact + n] = (__bf16)act;
  }
}

// ---------------------------------------------------------------------------
// RoPE + per-head RMSNorm -> bf16 Q/K/V in head-major layouts.
// grid = (S, 32 slots), block = 64.
// ---------------------------------------------------------------------------
__global__ __launch_bounds__(64)
void rope_norm_kernel(const float* __restrict__ qkv,
                      const float* __restrict__ cosb, const float* __restrict__ sinb,
                      const float* __restrict__ qw, const float* __restrict__ kw,
                      __bf16* __restrict__ Qo, __bf16* __restrict__ Ko,
                      __bf16* __restrict__ Vo) {
  const int t = blockIdx.x;
  const int slot = blockIdx.y;       // kvh*(G+2)+j
  const int kvh = slot >> 2;
  const int j = slot & 3;            // 0,1:q  2:k  3:v
  const int d = threadIdx.x;
  const float* src = qkv + (long)t * 2048 + slot * HD_;
  const float x = src[d];
  __shared__ float red[2];
  float val;
  if (j <= 2) {
    const float rot = (d < 32) ? -src[d + 32] : src[d - 32];
    val = x * cosb[t * HD_ + d] + rot * sinb[t * HD_ + d];
    float ss = val * val;
    for (int m = 16; m >= 1; m >>= 1) ss += __shfl_xor(ss, m, 32);
    if ((d & 31) == 0) red[d >> 5] = ss;
    __syncthreads();
    ss = red[0] + red[1];
    const float inv = rsqrtf(ss / (float)HD_ + 1e-6f);
    val = val * inv * ((j < 2) ? qw[d] : kw[d]);
  } else {
    val = x;
  }
  if (j < 2)       Qo[((long)(kvh * G_ + j) * S_ + t) * HD_ + d] = (__bf16)val;
  else if (j == 2) Ko[((long)kvh * S_ + t) * HD_ + d] = (__bf16)val;
  else             Vo[((long)kvh * S_ + t) * HD_ + d] = (__bf16)val;
}

// ---------------------------------------------------------------------------
// Flash attention with WMMA.  grid = (NH, S/64), block = 128 (4 waves, each
// owns 16 q rows). Q/K fragments straight from global; V staged transposed in
// LDS; P rearranged through per-wave LDS; online softmax via shfl per half.
// ---------------------------------------------------------------------------
__global__ __launch_bounds__(128)
void attn_kernel(const __bf16* __restrict__ Q, const __bf16* __restrict__ Kb,
                 const __bf16* __restrict__ Vb, __bf16* __restrict__ O) {
  const int h  = blockIdx.x;
  const int qb = blockIdx.y;
  const int kvh = h >> 1;                 // G = 2
  const int tid = threadIdx.x;
  const int lane = tid & 31, wid = tid >> 5;
  const bool hi = lane >= 16;
  const int l15 = lane & 15;
  const int q0 = qb * 64 + wid * 16;

  __shared__ __align__(16) __bf16 Vt[HD_][32];     // transposed V block
  __shared__ __align__(16) __bf16 Pb[4][16][32];   // per-wave P tiles

  Frag qa[2];
  {
    const char* qrow = (const char*)(Q + ((long)h * S_ + (q0 + l15)) * HD_);
    for (int c = 0; c < 2; ++c) {
      const int o = hi ? 16 : 0;
      qa[c].u[0] = *(const u32x4*)(qrow + c * 64 + o);
      qa[c].u[1] = *(const u32x4*)(qrow + c * 64 + 32 + o);
    }
  }

  float m_i[8], l_i[8];
  v8f o_acc[4] = {{}, {}, {}, {}};
  for (int r = 0; r < 8; ++r) { m_i[r] = -1e30f; l_i[r] = 0.f; }

  const int kmax = qb * 64 + 64;
  for (int kb = 0; kb < kmax; kb += 32) {
    {
      const int r = tid >> 2;             // kv row 0..31
      const int db = (tid & 3) * 16;      // d base
      const __bf16* vrow = Vb + ((long)kvh * S_ + (kb + r)) * HD_ + db;
      for (int i = 0; i < 16; ++i) Vt[db + i][r] = vrow[i];
    }
    __syncthreads();

    v8f sc[2];
    for (int nt = 0; nt < 2; ++nt) {
      v8f s = {};
      const char* krow = (const char*)(Kb + ((long)kvh * S_ + (kb + nt * 16 + l15)) * HD_);
      for (int c = 0; c < 2; ++c) {
        const v16bf kf = load_fragB(krow + c * 64, hi);
        s = __builtin_amdgcn_wmma_f32_16x16x32_bf16(false, qa[c].v, false, kf,
                                                    (short)0, s, false, false);
      }
      sc[nt] = s;
    }

    for (int r = 0; r < 8; ++r) {
      const int qrow = q0 + r + (hi ? 8 : 0);
      float mx = -1e30f;
      for (int nt = 0; nt < 2; ++nt) {
        const int krow = kb + nt * 16 + l15;
        float v = sc[nt][r] * 0.125f;
        v = (krow <= qrow) ? v : -1e30f;
        sc[nt][r] = v;
        mx = fmaxf(mx, v);
      }
      for (int m = 8; m >= 1; m >>= 1) mx = fmaxf(mx, __shfl_xor(mx, m, 32));
      const float newm = fmaxf(m_i[r], mx);
      const float corr = __expf(m_i[r] - newm);
      float rs = 0.f;
      for (int nt = 0; nt < 2; ++nt) {
        const float p = __expf(sc[nt][r] - newm);
        sc[nt][r] = p;
        rs += p;
      }
      for (int m = 8; m >= 1; m >>= 1) rs += __shfl_xor(rs, m, 32);
      l_i[r] = l_i[r] * corr + rs;
      m_i[r] = newm;
      for (int t4 = 0; t4 < 4; ++t4) o_acc[t4][r] *= corr;
    }

    for (int r = 0; r < 8; ++r) {
      const int mrow = r + (hi ? 8 : 0);
      Pb[wid][mrow][l15]      = (__bf16)sc[0][r];
      Pb[wid][mrow][16 + l15] = (__bf16)sc[1][r];
    }
    const v16bf pf = load_fragA((const char*)&Pb[wid][l15][0], hi);

    for (int t4 = 0; t4 < 4; ++t4) {
      const v16bf vf = load_fragB((const char*)&Vt[t4 * 16 + l15][0], hi);
      o_acc[t4] = __builtin_amdgcn_wmma_f32_16x16x32_bf16(false, pf, false, vf,
                                                          (short)0, o_acc[t4],
                                                          false, false);
    }
    __syncthreads();
  }

  for (int r = 0; r < 8; ++r) {
    const int qrow = q0 + r + (hi ? 8 : 0);
    const float inv = 1.f / l_i[r];
    for (int t4 = 0; t4 < 4; ++t4)
      O[(long)qrow * (NH_ * HD_) + h * HD_ + t4 * 16 + l15] =
          (__bf16)(o_acc[t4][r] * inv);
  }
}

// ---------------------------------------------------------------------------
// MoE gate: logits -> softmax -> top-8 -> renormalize.  grid=S, block=64.
// ---------------------------------------------------------------------------
__global__ __launch_bounds__(64)
void gate_topk_kernel(const float* __restrict__ x2f, const float* __restrict__ Wg,
                      float* __restrict__ w8, int* __restrict__ idx8) {
  const int t = blockIdx.x, e = threadIdx.x;
  __shared__ float xs[H_];
  __shared__ float gl[E_];
  for (int i = e; i < H_; i += 64) xs[i] = x2f[(long)t * H_ + i];
  __syncthreads();
  float acc = 0.f;
  for (int hh = 0; hh < H_; ++hh) acc += xs[hh] * Wg[(long)hh * E_ + e];
  gl[e] = acc;
  __syncthreads();
  if (e == 0) {
    float mx = gl[0];
    for (int i = 1; i < E_; ++i) mx = fmaxf(mx, gl[i]);
    float sum = 0.f;
    for (int i = 0; i < E_; ++i) { gl[i] = __expf(gl[i] - mx); sum += gl[i]; }
    float wv[TOPK_]; int id[TOPK_]; float wsum = 0.f;
    for (int k = 0; k < TOPK_; ++k) {
      float best = -1.f; int bi = 0;
      for (int i = 0; i < E_; ++i)
        if (gl[i] > best) { best = gl[i]; bi = i; }
      wv[k] = best / sum; id[k] = bi; gl[bi] = -1.f; wsum += wv[k];
    }
    const float denom = fmaxf(wsum, 1.1920929e-07f);
    for (int k = 0; k < TOPK_; ++k) {
      w8[t * TOPK_ + k] = wv[k] / denom;
      idx8[t * TOPK_ + k] = id[k];
    }
  }
}

// ---------------------------------------------------------------------------
// Routing tables: per-expert sequential scan (token-major cumsum semantics,
// fully deterministic). 1 block of 64 threads, one per expert.
// ---------------------------------------------------------------------------
__global__ __launch_bounds__(64)
void route_kernel(const int* __restrict__ idx8, const float* __restrict__ w8,
                  int* __restrict__ cnt, int* __restrict__ slot_tok,
                  int* __restrict__ slot_asgn, float* __restrict__ wv) {
  const int e = threadIdx.x;
  int c = 0;
  for (int r = 0; r < NASGN; ++r) {
    if (idx8[r] == e) {
      const bool valid = c < CAP_;
      if (valid) {
        slot_tok[e * CAP_ + c]  = r >> 3;   // token id
        slot_asgn[e * CAP_ + c] = r;        // assignment row
      }
      wv[r] = valid ? w8[r] : 0.f;
      ++c;
    }
  }
  cnt[e] = (c < CAP_) ? c : CAP_;
}

// ---------------------------------------------------------------------------
// Final combine: out = h1 + sharedDown + sum_k wv * y[assignment].
// Deterministic (per-token gather, no atomics). grid=S, block=256.
// ---------------------------------------------------------------------------
__global__ __launch_bounds__(256)
void combine_kernel(const float* __restrict__ h1, const float* __restrict__ sd,
                    const float* __restrict__ y, const float* __restrict__ wv,
                    float* __restrict__ out) {
  const int t = blockIdx.x;
  const int i = threadIdx.x * 4;
  const float4 a = *(const float4*)(h1 + (long)t * H_ + i);
  const float4 b = *(const float4*)(sd + (long)t * H_ + i);
  float4 acc = {a.x + b.x, a.y + b.y, a.z + b.z, a.w + b.w};
  for (int k = 0; k < TOPK_; ++k) {
    const float w = wv[t * TOPK_ + k];
    if (w != 0.f) {
      const float4 yv = *(const float4*)(y + (long)(t * TOPK_ + k) * H_ + i);
      acc.x += w * yv.x; acc.y += w * yv.y;
      acc.z += w * yv.z; acc.w += w * yv.w;
    }
  }
  *(float4*)(out + (long)t * H_ + i) = acc;
}

// ---------------------------------------------------------------------------
extern "C" void kernel_launch(void* const* d_in, const int* in_sizes, int n_in,
                              void* d_out, int out_size, void* d_ws, size_t ws_size,
                              hipStream_t stream) {
  const float* hidden = (const float*)d_in[0];
  const float* cosb   = (const float*)d_in[1];
  const float* sinb   = (const float*)d_in[2];
  const float* ln1w   = (const float*)d_in[3];
  const float* ln2w   = (const float*)d_in[4];
  const float* Wqkv   = (const float*)d_in[5];
  const float* Wo     = (const float*)d_in[6];
  const float* qnw    = (const float*)d_in[7];
  const float* knw    = (const float*)d_in[8];
  const float* WguS   = (const float*)d_in[9];
  const float* WdS    = (const float*)d_in[10];
  const float* Wgate  = (const float*)d_in[11];
  const float* WguE   = (const float*)d_in[12];
  const float* WdE    = (const float*)d_in[13];
  float* out = (float*)d_out;

  char* ws = (char*)d_ws;
  size_t off = 0;
  auto carve = [&](size_t bytes) -> char* {
    char* p = ws + off;
    off = (off + bytes + 255) & ~(size_t)255;
    return p;
  };
  __bf16* xbf     = (__bf16*)carve((size_t)S_ * H_ * 2);
  float*  qkv     = (float*) carve((size_t)S_ * 2048 * 4);
  __bf16* Qbf     = (__bf16*)carve((size_t)NH_ * S_ * HD_ * 2);
  __bf16* Kbf     = (__bf16*)carve((size_t)KVH_ * S_ * HD_ * 2);
  __bf16* Vbf     = (__bf16*)carve((size_t)KVH_ * S_ * HD_ * 2);
  __bf16* attnbf  = (__bf16*)carve((size_t)S_ * H_ * 2);
  float*  h1      = (float*) carve((size_t)S_ * H_ * 4);
  __bf16* x2bf    = (__bf16*)carve((size_t)S_ * H_ * 2);
  float*  x2f     = (float*) carve((size_t)S_ * H_ * 4);
  __bf16* act_sh  = (__bf16*)carve((size_t)S_ * I_ * 2);
  float*  shdown  = (float*) carve((size_t)S_ * H_ * 4);
  float*  w8      = (float*) carve((size_t)NASGN * 4);
  int*    idx8    = (int*)   carve((size_t)NASGN * 4);
  float*  wv      = (float*) carve((size_t)NASGN * 4);
  int*    cnt     = (int*)   carve((size_t)E_ * 4);
  int*    slot_tok  = (int*) carve((size_t)E_ * CAP_ * 4);
  int*    slot_asgn = (int*) carve((size_t)E_ * CAP_ * 4);
  __bf16* act_moe = (__bf16*)carve((size_t)MOE_ROWS * I_ * 2);
  float*  y_moe   = (float*) carve((size_t)MOE_ROWS * H_ * 4);
  (void)ws_size; (void)in_sizes; (void)n_in; (void)out_size;

  // 1) x = rmsnorm(hidden, ln1)  -> bf16
  rmsnorm_kernel<<<S_, 256, 0, stream>>>(hidden, ln1w, xbf, nullptr);

  // 2) qkv = x @ Wqkv   (1024 x 2048 x 1024)
  gemm_bf16<<<dim3(2048 / 32, S_ / 64, 1), 256, 0, stream>>>(
      xbf, H_, Wqkv, 2048, 0, qkv, 2048, H_,
      nullptr, nullptr, nullptr, nullptr, 0, 0);

  // 3) RoPE + q/k RMSNorm -> Q/K/V bf16
  rope_norm_kernel<<<dim3(S_, 32), 64, 0, stream>>>(
      qkv, cosb, sinb, qnw, knw, Qbf, Kbf, Vbf);

  // 4) flash attention (causal, GQA)
  attn_kernel<<<dim3(NH_, S_ / 64), 128, 0, stream>>>(Qbf, Kbf, Vbf, attnbf);

  // 5) h1 = hidden + attn @ Wo
  gemm_bf16<<<dim3(H_ / 32, S_ / 64, 1), 256, 0, stream>>>(
      attnbf, H_, Wo, H_, 0, h1, H_, H_,
      hidden, nullptr, nullptr, nullptr, 0, 0);

  // 6) x2 = rmsnorm(h1, ln2) -> bf16 + f32
  rmsnorm_kernel<<<S_, 256, 0, stream>>>(h1, ln2w, x2bf, x2f);

  // 7) shared SwiGLU up: act_sh = a * silu(b)
  ffn1_swiglu<<<dim3(I_ / 32, S_ / 64, 1), 256, 0, stream>>>(
      x2bf, H_, WguS, 0, 2 * I_, act_sh, I_, H_,
      nullptr, nullptr, nullptr, 0, 0);

  // 8) shared down: shdown = act_sh @ Wd_shared
  gemm_bf16<<<dim3(H_ / 32, S_ / 64, 1), 256, 0, stream>>>(
      act_sh, I_, WdS, H_, 0, shdown, H_, I_,
      nullptr, nullptr, nullptr, nullptr, 0, 0);

  // 9) gate + top-8
  gate_topk_kernel<<<S_, 64, 0, stream>>>(x2f, Wgate, w8, idx8);

  // 10) routing tables
  route_kernel<<<1, 64, 0, stream>>>(idx8, w8, cnt, slot_tok, slot_asgn, wv);

  // 11) MoE FFN1 (gathered rows, fused SwiGLU, scattered to assignment rows)
  ffn1_swiglu<<<dim3(I_ / 32, CAP_ / 64, E_), 256, 0, stream>>>(
      x2bf, H_, WguE, (long)H_ * 2 * I_, 2 * I_, act_moe, I_, H_,
      slot_tok, slot_asgn, cnt, CAP_, DUMMY_BASE);

  // 12) MoE FFN2 (down-proj, gathered/scattered by assignment rows)
  gemm_bf16<<<dim3(H_ / 32, CAP_ / 64, E_), 256, 0, stream>>>(
      act_moe, I_, WdE, H_, (long)I_ * H_, y_moe, H_, I_,
      nullptr, slot_asgn, slot_asgn, cnt, CAP_, DUMMY_BASE);

  // 13) out = h1 + shared + moe (deterministic per-token gather)
  combine_kernel<<<S_, 256, 0, stream>>>(h1, shdown, y_moe, wv, out);
}
